// STGCNFeatureExtractor_33818572489275
// MI455X (gfx1250) — compile-verified, matrix operand order checked by reading, not run on hardware
//
#include <hip/hip_runtime.h>
#include <hip/hip_bf16.h>

// MI455X / gfx1250 implementation.
// Strategy: dense 64x64 normalized adjacency (graph identical per frame),
// layer1 = cheap VALU, layer2 = f16 WMMA (v_wmma_f32_16x16x32_f16) with f32
// accumulation, mean fused into layer2 epilogue. H1 kept as f16 (134 MB,
// fits in the 192 MB L2) so layer2 B-operands stay on-chip.

typedef __attribute__((ext_vector_type(16))) _Float16 v16h;
typedef __attribute__((ext_vector_type(8)))  _Float16 h8;
typedef __attribute__((ext_vector_type(8)))  float    v8f;

union V16 { v16h v; h8 h[2]; };
union V8F { v8f v; float f[8]; };

#define NN   64     // nodes
#define NF   4096   // frames
#define IC   9      // input channels
#define H1D  256
#define H2D  512
#define NC   512
#define NE   256    // edges

// Workspace layout (bytes). Total ~151 MB.
#define WS_A32   0u           // 64*64*4          = 16384
#define WS_A16   16384u       // 64*64*2          = 8192
#define WS_W2T   32768u       // 512*256*2        = 262144
#define WS_PART  294912u      // 4096*512*4       = 8388608
#define WS_FEAT  8683520u     // 512*4
#define WS_H1T   16777216u    // 4096*256*64*2    = 134217728 (f16, [f][ch][node])

// ---------------------------------------------------------------- kernel 0
// Build A_hat: A[d][s] += dis[s]*dis[d] per edge, + dis[i]^2 on diagonal.
__global__ void k_build_ahat(const long long* __restrict__ ei,
                             float* __restrict__ A32, _Float16* __restrict__ A16) {
    __shared__ float Asm[NN * NN];
    __shared__ float dis[NN];
    int t = threadIdx.x;
    for (int i = t; i < NN * NN; i += 256) Asm[i] = 0.f;
    if (t < NN) {
        float deg = 1.f;  // self-loop
        for (int e = 0; e < NE; ++e)
            if ((int)ei[NE + e] == t) deg += 1.f;
        dis[t] = rsqrtf(deg);
    }
    __syncthreads();
    if (t < NE) {
        int s = (int)ei[t];
        int d = (int)ei[NE + t];
        atomicAdd(&Asm[d * NN + s], dis[s] * dis[d]);
    }
    if (t < NN) atomicAdd(&Asm[t * NN + t], dis[t] * dis[t]);
    __syncthreads();
    for (int i = t; i < NN * NN; i += 256) {
        float v = Asm[i];
        A32[i] = v;
        A16[i] = (_Float16)v;
    }
}

// ---------------------------------------------------------------- kernel 1
// W2 [256][512] fp32 -> W2t [512][256] f16 (so WMMA B-frags are contiguous).
__global__ void k_w2t(const float* __restrict__ W2, _Float16* __restrict__ W2t) {
    int i = blockIdx.x * 256 + threadIdx.x;   // i over 512*256
    int n = i >> 8, k = i & 255;
    W2t[i] = (_Float16)W2[k * H2D + n];
}

// ---------------------------------------------------------------- kernel 2
// Layer 1, one frame per block: G1 = A_hat @ Xf (64x9), H1 = relu(G1@W1+b1),
// stored transposed as f16: H1t[f][ch][node].
__global__ void k_layer1(const float* __restrict__ x, const float* __restrict__ A32,
                         const float* __restrict__ W1, const float* __restrict__ b1,
                         _Float16* __restrict__ H1t) {
    __shared__ float As[NN * NN];
    __shared__ float Xs[NN][IC];
    __shared__ float Gs[NN][IC + 1];
    int f = blockIdx.x, t = threadIdx.x;
    for (int i = t; i < NN * NN; i += 256) As[i] = A32[i];
    for (int i = t; i < NN * IC; i += 256) {
        int node = i / IC, ch = i % IC;
        Xs[node][ch] = x[((size_t)node * NF + f) * IC + ch];
    }
    __syncthreads();
    for (int i = t; i < NN * IC; i += 256) {
        int node = i / IC, ch = i % IC;
        float s = 0.f;
        #pragma unroll 8
        for (int j = 0; j < NN; ++j) s += As[node * NN + j] * Xs[j][ch];
        Gs[node][ch] = s;
    }
    __syncthreads();
    _Float16* outp = H1t + (size_t)f * H1D * NN;
    for (int i = t; i < NN * H1D; i += 256) {
        int c = i >> 6, node = i & 63;      // consecutive threads -> consecutive nodes
        float s = b1[c];
        #pragma unroll
        for (int ch = 0; ch < IC; ++ch) s += Gs[node][ch] * W1[ch * H1D + c];
        outp[c * NN + node] = (_Float16)fmaxf(s, 0.f);
    }
}

// ---------------------------------------------------------------- kernel 3
// Layer 2 WMMA workhorse. One frame per block, 512 threads = 16 wave32s.
// GEMM1: G[64x256] = Ahat16[64x64] @ H1f[64x256]   (B from H1t, transposed layout)
// GEMM2: H2[64x512] = G[64x256] @ W2               (B from W2t [512][256])
// Epilogue: relu(+bias), per-column sums -> part[f][512].
__global__ void __launch_bounds__(512, 1)
k_layer2(const _Float16* __restrict__ H1t, const _Float16* __restrict__ A16,
         const _Float16* __restrict__ W2t, const float* __restrict__ b2,
         float* __restrict__ part) {
    __shared__ _Float16 Gs[NN * 264];   // [node][ch], pitch 264 halves avoids bank conflicts
    __shared__ float colsum[H2D];

    int f    = blockIdx.x;
    int t    = threadIdx.x;
    int lane = t & 31;
    int w    = t >> 5;        // wave 0..15
    int m    = w & 3;         // M-tile (16 rows each)
    int g    = w >> 2;        // N-group 0..3
    int ln   = lane & 15;     // row/col within tile
    int base = (lane >> 4) * 8;   // K-half select per 16-bit fragment layout
    int half = lane >> 4;

    for (int i = t; i < H2D; i += 512) colsum[i] = 0.f;

    const _Float16* H1f = H1t + (size_t)f * H1D * NN;

    // ---- GEMM1: each wave computes 4 C-tiles (16x16) of G
    {
        V8F acc[4];
        #pragma unroll
        for (int j = 0; j < 4; ++j) acc[j].v = {};
        #pragma unroll
        for (int ks = 0; ks < 2; ++ks) {
            int kk = ks * 32;
            int arow = m * 16 + ln;
            V16 a;
            a.h[0] = *(const h8*)(A16 + arow * NN + kk + base);
            a.h[1] = *(const h8*)(A16 + arow * NN + kk + base + 16);
            #pragma unroll
            for (int j = 0; j < 4; ++j) {
                int n = (g * 4 + j) * 16 + ln;   // channel
                V16 b;
                b.h[0] = *(const h8*)(H1f + n * NN + kk + base);
                b.h[1] = *(const h8*)(H1f + n * NN + kk + base + 16);
                acc[j].v = __builtin_amdgcn_wmma_f32_16x16x32_f16(
                    false, a.v, false, b.v, (short)0, acc[j].v, false, false);
            }
        }
        // scatter G tiles to LDS as f16 [node][ch]
        #pragma unroll
        for (int j = 0; j < 4; ++j) {
            int ch = (g * 4 + j) * 16 + ln;
            #pragma unroll
            for (int r = 0; r < 8; ++r) {
                int node = m * 16 + half * 8 + r;   // D-frag: vgpr r, lane half -> M
                Gs[node * 264 + ch] = (_Float16)acc[j].f[r];
            }
        }
    }
    __syncthreads();

    // ---- GEMM2: each wave computes 8 C-tiles (16 rows x 128 cols)
    {
        V8F acc[8];
        #pragma unroll
        for (int j = 0; j < 8; ++j) acc[j].v = {};
        for (int ks = 0; ks < 8; ++ks) {
            int kk = ks * 32;
            int arow = m * 16 + ln;
            V16 a;
            a.h[0] = *(const h8*)(&Gs[arow * 264 + kk + base]);
            a.h[1] = *(const h8*)(&Gs[arow * 264 + kk + base + 16]);
            #pragma unroll
            for (int j = 0; j < 8; ++j) {
                int n = (g * 8 + j) * 16 + ln;
                V16 b;
                b.h[0] = *(const h8*)(W2t + n * H1D + kk + base);
                b.h[1] = *(const h8*)(W2t + n * H1D + kk + base + 16);
                acc[j].v = __builtin_amdgcn_wmma_f32_16x16x32_f16(
                    false, a.v, false, b.v, (short)0, acc[j].v, false, false);
            }
        }
        // bias + relu + column sums (mean fused)
        #pragma unroll
        for (int j = 0; j < 8; ++j) {
            int n = (g * 8 + j) * 16 + ln;
            float bb = b2[n];
            float s = 0.f;
            #pragma unroll
            for (int r = 0; r < 8; ++r) s += fmaxf(acc[j].f[r] + bb, 0.f);
            atomicAdd(&colsum[n], s);   // ds_add_f32
        }
    }
    __syncthreads();
    for (int i = t; i < H2D; i += 512) part[(size_t)f * H2D + i] = colsum[i];
}

// ---------------------------------------------------------------- kernel 4
__global__ void k_reduce(const float* __restrict__ part, float* __restrict__ feat) {
    int c = blockIdx.x * 256 + threadIdx.x;   // 2 blocks x 256 = 512 columns
    float s = 0.f;
    for (int fi = 0; fi < NF; ++fi) s += part[(size_t)fi * H2D + c];
    feat[c] = s * (1.0f / ((float)NN * (float)NF));
}

// ---------------------------------------------------------------- kernel 5
__global__ void k_fc(const float* __restrict__ feat, const float* __restrict__ Wfc,
                     const float* __restrict__ bfc, float* __restrict__ outp) {
    __shared__ float fs[H2D];
    int t = threadIdx.x;
    for (int i = t; i < H2D; i += 256) fs[i] = feat[i];
    __syncthreads();
    for (int k = t; k < NC; k += 256) {
        float s = bfc[k];
        for (int c = 0; c < H2D; ++c) s += fs[c] * Wfc[c * NC + k];
        outp[k] = s;
    }
}

extern "C" void kernel_launch(void* const* d_in, const int* in_sizes, int n_in,
                              void* d_out, int out_size, void* d_ws, size_t ws_size,
                              hipStream_t stream) {
    const float*     x   = (const float*)d_in[0];
    const long long* ei  = (const long long*)d_in[1];   // int64 edge_index [2][256]
    const float*     W1  = (const float*)d_in[2];
    const float*     b1  = (const float*)d_in[3];
    const float*     W2  = (const float*)d_in[4];
    const float*     b2  = (const float*)d_in[5];
    const float*     Wfc = (const float*)d_in[6];
    const float*     bfc = (const float*)d_in[7];
    float* outp = (float*)d_out;

    char* ws = (char*)d_ws;
    float*     A32  = (float*)(ws + WS_A32);
    _Float16*  A16  = (_Float16*)(ws + WS_A16);
    _Float16*  W2t  = (_Float16*)(ws + WS_W2T);
    float*     part = (float*)(ws + WS_PART);
    float*     feat = (float*)(ws + WS_FEAT);
    _Float16*  H1t  = (_Float16*)(ws + WS_H1T);

    k_build_ahat<<<1,   256, 0, stream>>>(ei, A32, A16);
    k_w2t      <<<512,  256, 0, stream>>>(W2, W2t);
    k_layer1   <<<NF,   256, 0, stream>>>(x, A32, W1, b1, H1t);
    k_layer2   <<<NF,   512, 0, stream>>>(H1t, A16, W2t, b2, part);
    k_reduce   <<<2,    256, 0, stream>>>(part, feat);
    k_fc       <<<1,    256, 0, stream>>>(feat, Wfc, bfc, outp);
}
